// DetectionLoss_73727408603747
// MI455X (gfx1250) — matrix operand order repaired and checked by reading, not compiled, not run on hardware
//
#include <hip/hip_runtime.h>

// ---------------- problem constants ----------------
#define NB   32
#define NQ   900
#define NT   32
#define NNEG 10
#define RD   256
#define TD   512
#define PD   256
#define BQ   (NB * NQ)        // 28800
#define BT   (NB * NT)        // 1024
#define NREG (BT + NB * NNEG) // 1344
#define COUT ((size_t)BQ * BT)

typedef __attribute__((ext_vector_type(16))) __bf16 v16bf;
typedef __attribute__((ext_vector_type(8)))  __bf16 v8bf;
typedef __attribute__((ext_vector_type(8)))  float  v8f;

// ---------------- helpers ----------------
__device__ __forceinline__ float giou_f(float4 a, float4 b) {
    float area1 = (a.z - a.x) * (a.w - a.y);
    float area2 = (b.z - b.x) * (b.w - b.y);
    float lx = fmaxf(a.x, b.x), ly = fmaxf(a.y, b.y);
    float rx = fminf(a.z, b.z), ry = fminf(a.w, b.w);
    float iw = fmaxf(rx - lx, 0.f), ih = fmaxf(ry - ly, 0.f);
    float inter = iw * ih;
    float uni = area1 + area2 - inter;
    float iou = inter / uni;
    float ex = fminf(a.x, b.x), ey = fminf(a.y, b.y);
    float fx = fmaxf(a.z, b.z), fy = fmaxf(a.w, b.w);
    float enc = (fx - ex) * (fy - ey);
    return iou - (enc - uni) / enc;
}

// ---------------- K0: zero accumulators ----------------
__global__ void k_init(float* acc) {
    if (threadIdx.x < 8) acc[threadIdx.x] = 0.f;
}

// ---------------- K1: text projection + normalize -> fp32 [32,256] and bf16 expanded [1024,256]
__global__ __launch_bounds__(256) void k_textproj(const float* __restrict__ te,
                                                  const float* __restrict__ Wt,
                                                  const float* __restrict__ bt,
                                                  float* __restrict__ txtf,
                                                  __bf16* __restrict__ txtb) {
    int b = blockIdx.x;       // 0..31
    int t = threadIdx.x;      // 0..255 (output column)
    const float* e = te + b * TD;
    float s = bt[t];
    for (int k = 0; k < TD; ++k) s = fmaf(e[k], Wt[k * PD + t], s);
    __shared__ float red[256];
    red[t] = s * s; __syncthreads();
    for (int off = 128; off > 0; off >>= 1) { if (t < off) red[t] += red[t + off]; __syncthreads(); }
    float inv = 1.f / fmaxf(sqrtf(red[0]), 1e-12f);
    float o = s * inv;
    txtf[b * PD + t] = o;
    __bf16 ob = (__bf16)o;
    for (int r = 0; r < NT; ++r) txtb[((size_t)(b * NT + r)) * PD + t] = ob;
}

// ---------------- K2: region row normalize -> bf16 [28800,256]
__global__ __launch_bounds__(256) void k_regnorm(const float* __restrict__ rf,
                                                 __bf16* __restrict__ regn) {
    int wave = threadIdx.x >> 5, lane = threadIdx.x & 31;
    int row = blockIdx.x * 8 + wave;           // grid 3600
    const float4* p = (const float4*)(rf + (size_t)row * RD) + lane * 2;
    float4 a = p[0], c = p[1];
    float ss = a.x*a.x + a.y*a.y + a.z*a.z + a.w*a.w + c.x*c.x + c.y*c.y + c.z*c.z + c.w*c.w;
    for (int off = 16; off > 0; off >>= 1) ss += __shfl_xor(ss, off, 32);
    float inv = 1.f / fmaxf(sqrtf(ss), 1e-12f);
    __bf16* q = regn + (size_t)row * RD + lane * 8;
    q[0] = (__bf16)(a.x * inv); q[1] = (__bf16)(a.y * inv);
    q[2] = (__bf16)(a.z * inv); q[3] = (__bf16)(a.w * inv);
    q[4] = (__bf16)(c.x * inv); q[5] = (__bf16)(c.y * inv);
    q[6] = (__bf16)(c.z * inv); q[7] = (__bf16)(c.w * inv);
}

// ---------------- K3: fused cost volume (WMMA bf16 GEMM + bbox/giou epilogue)
// grid (225, 16), block 256 (8 waves). wave -> one 16-row tile x 64 cols.
__global__ __launch_bounds__(256) void k_cost(const float* __restrict__ bbox,
                                              const float* __restrict__ tgt,
                                              const __bf16* __restrict__ regn,
                                              const __bf16* __restrict__ txtb,
                                              float* __restrict__ out) {
    int wave = threadIdx.x >> 5;
    int lane = threadIdx.x & 31;
    int hi   = lane >> 4;          // 0/1 selects K sub-range
    int lm   = lane & 15;
    int m_base = (blockIdx.x * 8 + wave) * 16;   // 0..28784
    int n_base = blockIdx.y * 64;                // 0..960

    v8f acc0 = {}, acc1 = {}, acc2 = {}, acc3 = {};

    const __bf16* arow = regn + (size_t)(m_base + lm) * RD + hi * 16;
    const __bf16* b0 = txtb + (size_t)(n_base + 0 * 16 + lm) * PD + hi * 16;
    const __bf16* b1 = txtb + (size_t)(n_base + 1 * 16 + lm) * PD + hi * 16;
    const __bf16* b2 = txtb + (size_t)(n_base + 2 * 16 + lm) * PD + hi * 16;
    const __bf16* b3 = txtb + (size_t)(n_base + 3 * 16 + lm) * PD + hi * 16;

    for (int k0 = 0; k0 < RD; k0 += 32) {
        // A fragment: 16 contiguous K at k0 + 16*hi
        v16bf a;
        {
            const v8bf* p = (const v8bf*)(arow + k0);
            v8bf lo = p[0], hh = p[1];
            #pragma unroll
            for (int i = 0; i < 8; ++i) { a[i] = lo[i]; a[8 + i] = hh[i]; }
        }
        // B fragments: 16 contiguous K at k0 + 16*hi
        #define LOADB(dst, base)                                   \
            v16bf dst;                                             \
            {                                                      \
                const v8bf* p = (const v8bf*)((base) + k0);        \
                v8bf lo = p[0], hh = p[1];                         \
                _Pragma("unroll")                                  \
                for (int i = 0; i < 8; ++i) { dst[i] = lo[i]; dst[8 + i] = hh[i]; } \
            }
        LOADB(bb0, b0)
        LOADB(bb1, b1)
        LOADB(bb2, b2)
        LOADB(bb3, b3)
        #undef LOADB
        acc0 = __builtin_amdgcn_wmma_f32_16x16x32_bf16(false, a, false, bb0, (short)0, acc0, false, false);
        acc1 = __builtin_amdgcn_wmma_f32_16x16x32_bf16(false, a, false, bb1, (short)0, acc1, false, false);
        acc2 = __builtin_amdgcn_wmma_f32_16x16x32_bf16(false, a, false, bb2, (short)0, acc2, false, false);
        acc3 = __builtin_amdgcn_wmma_f32_16x16x32_bf16(false, a, false, bb3, (short)0, acc3, false, false);
    }

    // Epilogue: D element (vgpr r, lane) = (M = m_base + r + 8*hi, N = n_base + c*16 + lm)
    #pragma unroll
    for (int c = 0; c < 4; ++c) {
        const v8f& ac = (c == 0) ? acc0 : (c == 1) ? acc1 : (c == 2) ? acc2 : acc3;
        int n = n_base + c * 16 + lm;
        float4 tb = ((const float4*)tgt)[n];
        #pragma unroll
        for (int r = 0; r < 8; ++r) {
            int m = m_base + r + 8 * hi;
            float4 pb = ((const float4*)bbox)[m];
            float l1 = fabsf(pb.x - tb.x) + fabsf(pb.y - tb.y) +
                       fabsf(pb.z - tb.z) + fabsf(pb.w - tb.w);
            float g = giou_f(pb, tb);
            out[(size_t)m * BT + n] = -ac[r] + 5.0f * l1 - 2.0f * g;
        }
    }
}

// ---------------- K4: simU [32,1344] (unique sim rows, scaled by 1/TEMP)
__global__ __launch_bounds__(256) void k_sim(const float* __restrict__ rf,
                                             const int* __restrict__ pidx,
                                             const int* __restrict__ nidx,
                                             const float* __restrict__ txtf,
                                             float* __restrict__ simU) {
    int j = blockIdx.x;        // 0..1343: region column
    int t = threadIdx.x;
    int b, q;
    if (j < BT) { b = j >> 5; q = pidx[j]; }
    else { int jj = j - BT; b = jj / NNEG; q = nidx[b * NNEG + jj % NNEG]; }
    const float* row = rf + ((size_t)b * NQ + q) * RD;
    __shared__ float r[256];
    __shared__ float red[256];
    float v = row[t];
    red[t] = v * v; __syncthreads();
    for (int off = 128; off > 0; off >>= 1) { if (t < off) red[t] += red[t + off]; __syncthreads(); }
    float inv = 1.f / fmaxf(sqrtf(red[0]), 1e-12f);
    r[t] = v * inv; __syncthreads();
    int wave = t >> 5, lane = t & 31;
    for (int u = wave; u < NB; u += 8) {
        const float* tx = txtf + u * PD;
        float s = 0.f;
        for (int k = lane; k < RD; k += 32) s = fmaf(r[k], tx[k], s);
        for (int off = 16; off > 0; off >>= 1) s += __shfl_xor(s, off, 32);
        if (lane == 0) simU[u * NREG + j] = s * (1.0f / 0.07f);
    }
}

// ---------------- K5: per-unique-row logsumexp
__global__ __launch_bounds__(256) void k_lse(const float* __restrict__ simU,
                                             float* __restrict__ lse) {
    int u = blockIdx.x, t = threadIdx.x;
    __shared__ float red[256];
    float m = -1e30f;
    for (int j = t; j < NREG; j += 256) m = fmaxf(m, simU[u * NREG + j]);
    red[t] = m; __syncthreads();
    for (int off = 128; off > 0; off >>= 1) { if (t < off) red[t] = fmaxf(red[t], red[t + off]); __syncthreads(); }
    float mx = red[0]; __syncthreads();
    float s = 0.f;
    for (int j = t; j < NREG; j += 256) s += expf(simU[u * NREG + j] - mx);
    red[t] = s; __syncthreads();
    for (int off = 128; off > 0; off >>= 1) { if (t < off) red[t] += red[t + off]; __syncthreads(); }
    if (t == 0) lse[u] = mx + logf(red[0]);
}

// ---------------- K6a: softplus sum over cls_pred
__global__ __launch_bounds__(256) void k_cls(const float* __restrict__ cls, float* __restrict__ acc) {
    int i = blockIdx.x * 256 + threadIdx.x;
    float sp = 0.f;
    if (i < BQ) {
        float x = cls[i];
        sp = (x > 0.f) ? x + log1pf(expf(-x)) : log1pf(expf(x));
    }
    __shared__ float red[256];
    red[threadIdx.x] = sp; __syncthreads();
    for (int off = 128; off > 0; off >>= 1) { if (threadIdx.x < off) red[threadIdx.x] += red[threadIdx.x + off]; __syncthreads(); }
    if (threadIdx.x == 0) atomicAdd(acc + 0, red[0]);
}

// ---------------- K6b: matched sums (x, L1, giou)
__global__ __launch_bounds__(256) void k_matched(const float* __restrict__ cls,
                                                 const float* __restrict__ bbox,
                                                 const float* __restrict__ tgt,
                                                 const int* __restrict__ pidx,
                                                 float* __restrict__ acc) {
    int i = blockIdx.x * 256 + threadIdx.x;   // grid 4
    float xs = 0.f, l1s = 0.f, gs = 0.f;
    if (i < BT) {
        int b = i >> 5;
        int q = pidx[i];
        int m = b * NQ + q;
        xs = cls[m];
        float4 pb = ((const float4*)bbox)[m];
        float4 tb = ((const float4*)tgt)[i];
        l1s = fabsf(pb.x - tb.x) + fabsf(pb.y - tb.y) + fabsf(pb.z - tb.z) + fabsf(pb.w - tb.w);
        gs = giou_f(pb, tb);
    }
    __shared__ float r0[256], r1[256], r2[256];
    int t = threadIdx.x;
    r0[t] = xs; r1[t] = l1s; r2[t] = gs; __syncthreads();
    for (int off = 128; off > 0; off >>= 1) {
        if (t < off) { r0[t] += r0[t + off]; r1[t] += r1[t + off]; r2[t] += r2[t + off]; }
        __syncthreads();
    }
    if (t == 0) { atomicAdd(acc + 1, r0[0]); atomicAdd(acc + 2, r1[0]); atomicAdd(acc + 3, r2[0]); }
}

// ---------------- K7: finalize 4 losses
__global__ __launch_bounds__(256) void k_finalize(const float* __restrict__ acc,
                                                  const float* __restrict__ lse,
                                                  const float* __restrict__ simU,
                                                  float* __restrict__ outl) {
    __shared__ float red[256];
    int t = threadIdx.x;
    float d = 0.f;
    for (int i = t; i < BT; i += 256) d += simU[(i >> 5) * NREG + i];
    red[t] = d; __syncthreads();
    for (int off = 128; off > 0; off >>= 1) { if (t < off) red[t] += red[t + off]; __syncthreads(); }
    float diag = red[0]; __syncthreads();
    red[t] = (t < NB) ? lse[t] : 0.f; __syncthreads();
    for (int off = 128; off > 0; off >>= 1) { if (t < off) red[t] += red[t + off]; __syncthreads(); }
    if (t == 0) {
        float suml = red[0];
        outl[0] = 2.0f * (acc[0] - acc[1]) * (1.0f / (float)BQ);       // L_CLS * loss_cls
        outl[1] = 5.0f * acc[2] * (1.0f / (float)(BT * 4));            // L_BBOX * loss_l1
        outl[2] = 2.0f * (1.0f - acc[3] * (1.0f / (float)BT));         // L_GIOU * loss_giou
        outl[3] = (32.0f * suml - diag) * (1.0f / (float)BT);          // L_SIM * loss_sim
    }
}

// ---------------- launcher ----------------
extern "C" void kernel_launch(void* const* d_in, const int* in_sizes, int n_in,
                              void* d_out, int out_size, void* d_ws, size_t ws_size,
                              hipStream_t stream) {
    const float* rf   = (const float*)d_in[0];   // region_features [32,900,256]
    const float* bbox = (const float*)d_in[1];   // bbox_pred [32,900,4]
    const float* cls  = (const float*)d_in[2];   // cls_pred [32,900,1]
    const float* tgt  = (const float*)d_in[3];   // tgt_boxes [32,32,4]
    const float* te   = (const float*)d_in[4];   // text_embeddings [32,512]
    const int*   pidx = (const int*)d_in[5];     // pred_idx [32,32]
    const int*   nidx = (const int*)d_in[6];     // neg_idx [32,10]
    const float* Wt   = (const float*)d_in[7];   // [512,256]
    const float* bt   = (const float*)d_in[8];   // [256]
    float* out = (float*)d_out;

    char* ws = (char*)d_ws;
    float*  txtf = (float*)(ws + 0);             // 32*256*4      = 32768
    float*  acc  = (float*)(ws + 32768);         // 8*4 (pad 64)
    float*  lse  = (float*)(ws + 32832);         // 32*4 (pad)
    float*  simU = (float*)(ws + 33024);         // 32*1344*4     = 172032 -> 205056
    __bf16* txtb = (__bf16*)(ws + 205312);       // 1024*256*2    = 524288 -> 729600
    __bf16* regn = (__bf16*)(ws + 729856);       // 28800*256*2   = 14745600

    k_init<<<1, 32, 0, stream>>>(acc);
    k_textproj<<<NB, 256, 0, stream>>>(te, Wt, bt, txtf, txtb);
    k_regnorm<<<BQ / 8, 256, 0, stream>>>(rf, regn);
    k_cost<<<dim3(BQ / 16 / 8, BT / 64), 256, 0, stream>>>(bbox, tgt, regn, txtb, out);
    k_sim<<<NREG, 256, 0, stream>>>(rf, pidx, nidx, txtf, simU);
    k_lse<<<NB, 256, 0, stream>>>(simU, lse);
    k_cls<<<(BQ + 255) / 256, 256, 0, stream>>>(cls, acc);
    k_matched<<<(BT + 255) / 256, 256, 0, stream>>>(cls, bbox, tgt, pidx, acc);
    k_finalize<<<1, 256, 0, stream>>>(acc, lse, simU, out + COUT);
}
